// Complex_32160715113072
// MI455X (gfx1250) — matrix operand order repaired and checked by reading, not compiled, not run on hardware
//
#include <hip/hip_runtime.h>

#define N_ENT   50000
#define DIM     128
#define N_EDGES 400000
#define NB      4
#define COLS    640   // 4*128 basis-projected cols + 128 root cols

typedef __attribute__((ext_vector_type(16))) __bf16 v16bf;
typedef __attribute__((ext_vector_type(8)))  float  v8f;

union Frag16 { v16bf v; uint4 q[2]; };

__device__ __forceinline__ unsigned int f2bf_bits(float f) {
    unsigned int a = __float_as_uint(f);
    return (a + 0x7FFFu + ((a >> 16) & 1u)) >> 16;   // RNE f32->bf16
}
__device__ __forceinline__ unsigned int pack2bf(float lo, float hi) {
    return (f2bf_bits(lo) & 0xFFFFu) | (f2bf_bits(hi) << 16);
}
__device__ __forceinline__ float bflo(unsigned int u) { return __uint_as_float(u << 16); }
__device__ __forceinline__ float bfhi(unsigned int u) { return __uint_as_float(u & 0xFFFF0000u); }

// Pack W = [basis0|basis1|basis2|basis3|root] as bf16, column-major: wt[j*128 + k]
__global__ void prep_w_kernel(const float* __restrict__ basis, const float* __restrict__ root,
                              unsigned short* __restrict__ wt) {
    int t = blockIdx.x * blockDim.x + threadIdx.x;
    if (t >= COLS * DIM) return;
    int j = t >> 7;      // 0..639
    int k = t & 127;     // 0..127
    float v;
    if (j < NB * DIM) {
        int b = j >> 7, o = j & 127;
        v = basis[(b * DIM + k) * DIM + o];
    } else {
        v = root[k * DIM + (j - NB * DIM)];
    }
    wt[j * DIM + k] = (unsigned short)f2bf_bits(v);
}

// Gather rows (optional) + convert f32 -> bf16, 4 elements per thread.
__global__ void convert_kernel(const float* __restrict__ X, const int* __restrict__ rowidx,
                               unsigned short* __restrict__ Xbf) {
    int t = blockIdx.x * blockDim.x + threadIdx.x;      // over N_ENT*DIM/4
    if (t >= N_ENT * DIM / 4) return;
    int n = t >> 5;              // node
    int o = (t & 31) * 4;        // feature offset
    int row = rowidx ? rowidx[n] : n;
    float4 f = *(const float4*)(X + (size_t)row * DIM + o);
    uint2 u;
    u.x = pack2bf(f.x, f.y);
    u.y = pack2bf(f.z, f.w);
    *(uint2*)(Xbf + (size_t)n * DIM + o) = u;
}

// Y[n, 0:512] = nb (basis-projected x), Y[n, 512:640] = x @ root ; Y stored bf16.
// Each wave computes a 16(M) x 64(N) tile; K=128 in 4 steps of 32.
// Per K-step: 2 A-frag b128 loads + 8 B-frag b128 loads (single clause), then 4 WMMAs
// so the tensor ops overlap the remaining loads via partial loadcnt waits.
__global__ __launch_bounds__(256)
void gemm_wmma_kernel(const unsigned short* __restrict__ Xbf,
                      const unsigned short* __restrict__ wt,
                      unsigned short* __restrict__ Y) {
    const int wave = (blockIdx.x * blockDim.x + threadIdx.x) >> 5;
    const int lane = threadIdx.x & 31;
    const int MT = N_ENT / 16;    // 3125
    const int JT = COLS / 64;     // 10
    if (wave >= MT * JT) return;
    const int m0 = (wave / JT) * 16;
    const int j0 = (wave % JT) * 64;
    const int g  = lane >> 4;
    const int ln = lane & 15;

    const unsigned short* xr = Xbf + (size_t)(m0 + ln) * DIM;
    const unsigned short* wb = wt + (size_t)(j0 + ln) * DIM;   // base col for this lane

    v8f acc[4] = { {}, {}, {}, {} };

#pragma unroll
    for (int kk = 0; kk < 4; ++kk) {
        const int k0   = kk * 32 + g * 8;
        const int krow = kk * 32 + g * 8;

        Frag16 a;
        a.q[0] = *(const uint4*)(xr + k0);        // K = k0..k0+7
        a.q[1] = *(const uint4*)(xr + k0 + 16);   // K = k0+16..k0+23

        Frag16 b[4];
#pragma unroll
        for (int t = 0; t < 4; ++t) {
            const uint4* bp = (const uint4*)(wb + t * 16 * DIM + krow);
            b[t].q[0] = bp[0];
            b[t].q[1] = bp[2];
        }
#pragma unroll
        for (int t = 0; t < 4; ++t) {
            acc[t] = __builtin_amdgcn_wmma_f32_16x16x32_bf16(
                false, a.v, false, b[t].v, (short)0, acc[t], false, false);
        }
    }

#pragma unroll
    for (int t = 0; t < 4; ++t) {
        const int cc = j0 + t * 16 + ln;
#pragma unroll
        for (int r = 0; r < 8; ++r) {
            const int rr = m0 + r + g * 8;
            Y[(size_t)rr * COLS + cc] = (unsigned short)f2bf_bits(acc[t][r]);
        }
    }
}

__global__ void count_kernel(const int* __restrict__ dst, float* __restrict__ cnt) {
    int e = blockIdx.x * blockDim.x + threadIdx.x;
    if (e < N_EDGES) unsafeAtomicAdd(&cnt[dst[e]], 1.0f);
}

// One wave per edge; lane handles 4 output features; both channels fused.
__global__ __launch_bounds__(256)
void edge_kernel(const int* __restrict__ src, const int* __restrict__ dst,
                 const int* __restrict__ et, const float* __restrict__ enorm,
                 const float* __restrict__ att,
                 const unsigned short* __restrict__ Yr, const unsigned short* __restrict__ Yi,
                 float* __restrict__ aggr, float* __restrict__ aggi) {
    const int e = (blockIdx.x * blockDim.x + threadIdx.x) >> 5;
    if (e >= N_EDGES) return;
    const int lane = threadIdx.x & 31;
    const int s = src[e];
    const int d = dst[e];
    const int r = et[e];
    const float en = enorm[e];
    const float c0 = att[r * NB + 0], c1 = att[r * NB + 1];
    const float c2 = att[r * NB + 2], c3 = att[r * NB + 3];
    const int o = lane * 4;
    const unsigned short* br = Yr + (size_t)s * COLS;
    const unsigned short* bi = Yi + (size_t)s * COLS;

    float mr[4] = {0.f, 0.f, 0.f, 0.f};
    float mi[4] = {0.f, 0.f, 0.f, 0.f};
    const float cf[NB] = {c0, c1, c2, c3};
#pragma unroll
    for (int b = 0; b < NB; ++b) {
        uint2 pr = *(const uint2*)(br + b * DIM + o);
        uint2 pi = *(const uint2*)(bi + b * DIM + o);
        float c = cf[b];
        mr[0] += c * bflo(pr.x); mr[1] += c * bfhi(pr.x);
        mr[2] += c * bflo(pr.y); mr[3] += c * bfhi(pr.y);
        mi[0] += c * bflo(pi.x); mi[1] += c * bfhi(pi.x);
        mi[2] += c * bflo(pi.y); mi[3] += c * bfhi(pi.y);
    }
    float* ar = aggr + (size_t)d * DIM + o;
    float* ai = aggi + (size_t)d * DIM + o;
#pragma unroll
    for (int q = 0; q < 4; ++q) {
        unsafeAtomicAdd(ar + q, mr[q]);
        unsafeAtomicAdd(ai + q, mi[q] * en);
    }
}

// out = agg / max(cnt,1) + root_part(Y cols 512..639) + bias ; optional relu
// Writes f32 (outf) or bf16 (outbf) depending on which pointer is non-null.
__global__ void finalize_kernel(const float* __restrict__ agg, const float* __restrict__ cnt,
                                const unsigned short* __restrict__ Y, const float* __restrict__ bias,
                                float* __restrict__ outf, unsigned short* __restrict__ outbf,
                                int relu) {
    int t = blockIdx.x * blockDim.x + threadIdx.x;
    if (t >= N_ENT * DIM) return;
    int n = t >> 7, o = t & 127;
    float c = cnt[n];
    c = c < 1.0f ? 1.0f : c;
    float rootv = __uint_as_float(((unsigned int)Y[(size_t)n * COLS + 512 + o]) << 16);
    float v = agg[t] / c + rootv + bias[o];
    if (relu) v = v > 0.f ? v : 0.f;
    if (outf) outf[t] = v;
    else      outbf[t] = (unsigned short)f2bf_bits(v);
}

extern "C" void kernel_launch(void* const* d_in, const int* in_sizes, int n_in,
                              void* d_out, int out_size, void* d_ws, size_t ws_size,
                              hipStream_t stream) {
    const int*   entity = (const int*)d_in[0];
    const int*   eidx   = (const int*)d_in[1];
    const int*   etype  = (const int*)d_in[2];
    const float* enorm  = (const float*)d_in[3];
    const float* emb_r  = (const float*)d_in[4];
    const float* emb_i  = (const float*)d_in[5];
    const float* basis1 = (const float*)d_in[6];
    const float* att1   = (const float*)d_in[7];
    const float* root1  = (const float*)d_in[8];
    const float* bias1  = (const float*)d_in[9];
    const float* basis2 = (const float*)d_in[10];
    const float* att2   = (const float*)d_in[11];
    const float* root2  = (const float*)d_in[12];
    const float* bias2  = (const float*)d_in[13];
    const int* src = eidx;
    const int* dst = eidx + N_EDGES;

    char* ws = (char*)d_ws;
    size_t off = 0;
    auto take = [&](size_t bytes) -> void* {
        void* p = ws + off;
        off += (bytes + 255) & ~(size_t)255;
        return p;
    };
    unsigned short* Yr   = (unsigned short*)take((size_t)N_ENT * COLS * 2);
    unsigned short* Yi   = (unsigned short*)take((size_t)N_ENT * COLS * 2);
    float* aggr = (float*)take((size_t)N_ENT * DIM * 4);
    float* aggi = (float*)take((size_t)N_ENT * DIM * 4);
    float* cnt  = (float*)take((size_t)N_ENT * 4);
    unsigned short* hrbf = (unsigned short*)take((size_t)N_ENT * DIM * 2);
    unsigned short* hibf = (unsigned short*)take((size_t)N_ENT * DIM * 2);
    unsigned short* xbf  = (unsigned short*)take((size_t)N_ENT * DIM * 2);
    unsigned short* wt   = (unsigned short*)take((size_t)COLS * DIM * 2);

    float* out_r = (float*)d_out;
    float* out_i = out_r + (size_t)N_ENT * DIM;

    const int gemmWaves  = (N_ENT / 16) * (COLS / 64);        // 31250
    const int gemmBlocks = (gemmWaves * 32 + 255) / 256;      // 3907
    const int edgeBlocks = (N_EDGES * 32 + 255) / 256;        // 50000
    const int nodeBlocks = (N_ENT * DIM + 255) / 256;         // 25000
    const int convBlocks = (N_ENT * DIM / 4 + 255) / 256;     // 6250

    // ---- shared count (graph identical for both layers/channels)
    hipMemsetAsync(cnt, 0, (size_t)N_ENT * 4, stream);
    count_kernel<<<(N_EDGES + 255) / 256, 256, 0, stream>>>(dst, cnt);

    // ================= Layer 1 =================
    hipMemsetAsync(aggr, 0, (size_t)N_ENT * DIM * 4, stream);
    hipMemsetAsync(aggi, 0, (size_t)N_ENT * DIM * 4, stream);
    prep_w_kernel<<<(COLS * DIM + 255) / 256, 256, 0, stream>>>(basis1, root1, wt);
    convert_kernel<<<convBlocks, 256, 0, stream>>>(emb_r, entity, xbf);
    gemm_wmma_kernel<<<gemmBlocks, 256, 0, stream>>>(xbf, wt, Yr);
    convert_kernel<<<convBlocks, 256, 0, stream>>>(emb_i, entity, xbf);
    gemm_wmma_kernel<<<gemmBlocks, 256, 0, stream>>>(xbf, wt, Yi);
    edge_kernel<<<edgeBlocks, 256, 0, stream>>>(src, dst, etype, enorm, att1, Yr, Yi, aggr, aggi);
    finalize_kernel<<<nodeBlocks, 256, 0, stream>>>(aggr, cnt, Yr, bias1, nullptr, hrbf, 1);
    finalize_kernel<<<nodeBlocks, 256, 0, stream>>>(aggi, cnt, Yi, bias1, nullptr, hibf, 1);

    // ================= Layer 2 =================
    hipMemsetAsync(aggr, 0, (size_t)N_ENT * DIM * 4, stream);
    hipMemsetAsync(aggi, 0, (size_t)N_ENT * DIM * 4, stream);
    prep_w_kernel<<<(COLS * DIM + 255) / 256, 256, 0, stream>>>(basis2, root2, wt);
    gemm_wmma_kernel<<<gemmBlocks, 256, 0, stream>>>(hrbf, wt, Yr);
    gemm_wmma_kernel<<<gemmBlocks, 256, 0, stream>>>(hibf, wt, Yi);
    edge_kernel<<<edgeBlocks, 256, 0, stream>>>(src, dst, etype, enorm, att2, Yr, Yi, aggr, aggi);
    finalize_kernel<<<nodeBlocks, 256, 0, stream>>>(aggr, cnt, Yr, bias2, out_r, nullptr, 0);
    finalize_kernel<<<nodeBlocks, 256, 0, stream>>>(aggi, cnt, Yi, bias2, out_i, nullptr, 0);
}
